// BertAdapterCapsuleMask_71459665871060
// MI455X (gfx1250) — matrix operand order repaired and checked by reading, not compiled, not run on hardware
//
#include <hip/hip_runtime.h>

typedef __bf16 bf16_t;
typedef bf16_t v16bf __attribute__((ext_vector_type(16)));
typedef bf16_t v8bf  __attribute__((ext_vector_type(8)));
typedef float  v8f   __attribute__((ext_vector_type(8)));

#define BB 128
#define SS 128
#define HH 768
#define AA 2048
#define CC 3
#define NT 10
#define MM (BB * SS)     // 16384
#define SC (SS * CC)     // 384
#define LDP 40           // padded LDS row stride (halfwords) for bank spread

__device__ __forceinline__ unsigned short f2bf(float f) {
  unsigned u = __float_as_uint(f);
  u += 0x7FFFu + ((u >> 16) & 1u);           // round-to-nearest-even
  return (unsigned short)(u >> 16);
}
__device__ __forceinline__ float sigmoidf_(float z) {
  return 1.0f / (1.0f + __expf(-z));
}

// ---------------------------------------------------------------------------
// Batched transpose + f32 -> bf16 convert:  in [Bt][R][C] -> out [Bt][C][R]
// ---------------------------------------------------------------------------
__global__ void k_transpose_bf16(const float* __restrict__ in,
                                 unsigned short* __restrict__ out,
                                 int Bt, int R, int C) {
  long total = (long)Bt * R * C;
  for (long o = (long)blockIdx.x * blockDim.x + threadIdx.x; o < total;
       o += (long)gridDim.x * blockDim.x) {
    long rc  = (long)R * C;
    long b   = o / rc;
    long rem = o - b * rc;
    long c   = rem / R;
    long r   = rem - c * R;
    out[o] = f2bf(in[b * rc + r * C + c]);
  }
}

// ---------------------------------------------------------------------------
// sem = einsum('bsh,nhc->bnsc', x, sem_w) + sem_b ; squash over n per (b,s,c)
// writes bf16 semA in A-layout [n][b][i], i = s*3+c.  One thread per (b,s).
// ---------------------------------------------------------------------------
__global__ void k_sem_squash(const float* __restrict__ x,
                             const float* __restrict__ sem_w,
                             const float* __restrict__ sem_b,
                             unsigned short* __restrict__ semA) {
  int bs = blockIdx.x * blockDim.x + threadIdx.x;
  if (bs >= MM) return;
  const float* xr = x + (long)bs * HH;
  float val[NT][CC];
#pragma unroll
  for (int n = 0; n < NT; ++n)
#pragma unroll
    for (int c = 0; c < CC; ++c) val[n][c] = sem_b[n * CC + c];

  for (int k = 0; k < HH; ++k) {
    float xv = xr[k];
    const float* w = sem_w + (long)k * CC;   // + n*HH*CC
#pragma unroll
    for (int n = 0; n < NT; ++n) {
      const float* wn = w + (long)n * HH * CC;
      val[n][0] = fmaf(xv, wn[0], val[n][0]);
      val[n][1] = fmaf(xv, wn[1], val[n][1]);
      val[n][2] = fmaf(xv, wn[2], val[n][2]);
    }
  }
#pragma unroll
  for (int c = 0; c < CC; ++c) {
    float ss = 0.0f;
#pragma unroll
    for (int n = 0; n < NT; ++n) ss = fmaf(val[n][c], val[n][c], ss);
    float fac = __fsqrt_rn(ss) / (1.0f + ss);     // v * sqrt(sq)/(1+sq)
#pragma unroll
    for (int n = 0; n < NT; ++n) {
      semA[(long)n * (BB * SC) + (long)bs * CC + c] = f2bf(val[n][c] * fac);
    }
  }
}

// ---------------------------------------------------------------------------
// priors[c][b][n][d] = sum_i semA[n][b][i] * rwT[c][n][d][i]   (bf16 WMMA)
// one wave per 16x16 output tile; 30 * 8 * 8 = 1920 tiles
// ---------------------------------------------------------------------------
__global__ void k_priors_wmma(const unsigned short* __restrict__ semA,
                              const unsigned short* __restrict__ rwT,
                              float* __restrict__ priors) {
  int lane   = threadIdx.x & 31;
  int tileId = blockIdx.x * (blockDim.x >> 5) + (threadIdx.x >> 5);
  int cn = tileId >> 6;           // 0..29
  int tm = (tileId >> 3) & 7;     // b-tile
  int tn = tileId & 7;            // d-tile
  int c  = cn / NT;
  int n  = cn % NT;

  const unsigned short* pA = semA + (long)n * (BB * SC);
  const unsigned short* pB = rwT + (long)cn * (SS * SC);
  int aRow = tm * 16 + (lane & 15);
  int bRow = tn * 16 + (lane & 15);
  int kA = (lane >> 4) << 3;      // 0 or 8
  int kB = (lane >> 4) << 3;      // 0 or 8

  v8f acc = {};
  for (int k0 = 0; k0 < SC; k0 += 32) {
    const unsigned short* ap = pA + (long)aRow * SC + k0 + kA;
    v8bf alo = *reinterpret_cast<const v8bf*>(ap);
    v8bf ahi = *reinterpret_cast<const v8bf*>(ap + 16);
    v16bf a = __builtin_shufflevector(alo, ahi, 0, 1, 2, 3, 4, 5, 6, 7, 8, 9,
                                      10, 11, 12, 13, 14, 15);
    const unsigned short* bp = pB + (long)bRow * SC + k0 + kB;
    v8bf blo = *reinterpret_cast<const v8bf*>(bp);
    v8bf bhi = *reinterpret_cast<const v8bf*>(bp + 16);
    v16bf b = __builtin_shufflevector(blo, bhi, 0, 1, 2, 3, 4, 5, 6, 7, 8, 9,
                                      10, 11, 12, 13, 14, 15);
    acc = __builtin_amdgcn_wmma_f32_16x16x32_bf16(false, a, false, b, (short)0,
                                                  acc, false, false);
  }
#pragma unroll
  for (int r = 0; r < 8; ++r) {
    int row = tm * 16 + r + ((lane >> 4) << 3);     // b
    int col = tn * 16 + (lane & 15);                // d
    priors[(((long)c * BB + row) * NT + n) * SS + col] = acc[r];
  }
}

// ---------------------------------------------------------------------------
// dynamic routing, 3 iterations; one block per (c,b), 128 threads (d)
// writes raw vote [C][B][S]
// ---------------------------------------------------------------------------
__global__ void k_routing(const float* __restrict__ priors,
                          const int* __restrict__ tptr,
                          float* __restrict__ vote) {
  int d  = threadIdx.x;
  int cb = blockIdx.x;            // c*128+b
  int t  = *tptr;
  __shared__ float red[SS];

  float pr[NT];
#pragma unroll
  for (int n = 0; n < NT; ++n)
    pr[n] = priors[((long)cb * NT + n) * SS + d];

  float logit[NT];
#pragma unroll
  for (int n = 0; n < NT; ++n) logit[n] = 0.0f;

  float vt = 0.0f;
  for (int it = 0; it < 3; ++it) {
    float mk[NT];
#pragma unroll
    for (int n = 0; n < NT; ++n) mk[n] = (n <= t) ? logit[n] : -10000.0f;
    float mx = mk[0];
#pragma unroll
    for (int n = 1; n < NT; ++n) mx = fmaxf(mx, mk[n]);
    float e[NT], sum = 0.0f;
#pragma unroll
    for (int n = 0; n < NT; ++n) { e[n] = __expf(mk[n] - mx); sum += e[n]; }
    float inv = 1.0f / sum;
    vt = 0.0f;
#pragma unroll
    for (int n = 0; n < NT; ++n) vt = fmaf(e[n] * inv, pr[n], vt);

    if (it < 2) {
      red[d] = vt * vt;
      __syncthreads();
      for (int off = SS / 2; off > 0; off >>= 1) {
        if (d < off) red[d] += red[d + off];
        __syncthreads();
      }
      float ss = red[0];
      __syncthreads();
      float outd = vt * (__fsqrt_rn(ss) / (1.0f + ss));   // squash(vote)
#pragma unroll
      for (int n = 0; n < NT; ++n) {
        red[d] = pr[n] * outd;
        __syncthreads();
        for (int off = SS / 2; off > 0; off >>= 1) {
          if (d < off) red[d] += red[d + off];
          __syncthreads();
        }
        logit[n] = mk[n] + red[0];
        __syncthreads();
      }
    }
  }
  vote[(long)cb * SS + d] = vt;
}

// ---------------------------------------------------------------------------
// h = x + (vote_flat @ larger_w + larger_b) * sigmoid(s*elarger[t]) -> bf16 A
// ---------------------------------------------------------------------------
__global__ void k_build_h(const float* __restrict__ x,
                          const float* __restrict__ vote,
                          const float* __restrict__ larger_w,
                          const float* __restrict__ larger_b,
                          const float* __restrict__ elarger,
                          const int* __restrict__ tptr,
                          const int* __restrict__ sptr,
                          unsigned short* __restrict__ hA) {
  long idx = (long)blockIdx.x * blockDim.x + threadIdx.x;
  if (idx >= (long)MM * HH) return;
  long bs = idx / HH;
  int  hc = (int)(idx - bs * HH);
  int   t  = *tptr;
  float sf = (float)(*sptr);
  float g  = sigmoidf_(sf * elarger[(long)t * HH + hc]);
  float acc = larger_b[hc];
#pragma unroll
  for (int c = 0; c < CC; ++c)
    acc = fmaf(vote[bs * CC + c], larger_w[c * HH + hc], acc);
  hA[idx] = f2bf(x[idx] + acc * g);
}

// ---------------------------------------------------------------------------
// Staged bf16 GEMM core.  A: [M][K] bf16 row-major, BT: [N][K] bf16.
// 256 threads = 8 waves (2x4); block tile 128x128; wave tile 64x32 (4x2 WMMA
// fragments).  K-step 32, double-buffered LDS staging filled with
// global_load_async_to_lds_b128 (ASYNCcnt) and consumed via ds_load_b128.
// ---------------------------------------------------------------------------
template <int KDIM>
__device__ __forceinline__ void gemm_staged(const unsigned short* __restrict__ A,
                                            const unsigned short* __restrict__ BT,
                                            v8f acc[4][2]) {
  __shared__ __align__(16) unsigned short As[2][128 * LDP];
  __shared__ __align__(16) unsigned short Bs[2][128 * LDP];

  int tid  = threadIdx.x;
  int lane = tid & 31;
  int wave = tid >> 5;
  int wm = wave >> 2, wn = wave & 3;
  int rowBlk = blockIdx.y * 128;
  int colBlk = blockIdx.x * 128;
  int kA = (lane >> 4) << 3;      // 0 or 8  (A frag K base, halfwords)
  int kB = (lane >> 4) << 3;      // 0 or 8  (B frag K base, halfwords)

  // each thread async-copies 2x16B of the A tile and 2x16B of the B tile
  int lr = tid >> 1;              // tile row/col 0..127
  int q0 = (tid & 1) << 1;        // 16B chunk index {0,2}

  const unsigned short* Ab = A + (long)rowBlk * KDIM;
  const unsigned short* Bb = BT + (long)colBlk * KDIM;

  auto issue = [&](const unsigned short* gbase, unsigned short* lds, int k0) {
#pragma unroll
    for (int j = 0; j < 2; ++j) {
      int q = q0 + j;
      const unsigned short* gp = gbase + (long)lr * KDIM + k0 + q * 8;
      unsigned ldsoff =
          (unsigned)(unsigned long long)(lds + lr * LDP + q * 8);
      asm volatile("global_load_async_to_lds_b128 %0, %1, off"
                   :
                   : "v"(ldsoff), "v"(gp)
                   : "memory");
    }
  };

  issue(Ab, &As[0][0], 0);
  issue(Bb, &Bs[0][0], 0);
  asm volatile("s_wait_asynccnt 0" ::: "memory");
  __syncthreads();

  constexpr int STEPS = KDIM / 32;
  for (int kk = 0; kk < STEPS; ++kk) {
    int buf = kk & 1;
    if (kk + 1 < STEPS) {
      issue(Ab, &As[buf ^ 1][0], (kk + 1) * 32);
      issue(Bb, &Bs[buf ^ 1][0], (kk + 1) * 32);
    }
    v16bf afr[4];
#pragma unroll
    for (int mf = 0; mf < 4; ++mf) {
      const unsigned short* p =
          &As[buf][(wm * 64 + mf * 16 + (lane & 15)) * LDP + kA];
      v8bf lo = *reinterpret_cast<const v8bf*>(p);
      v8bf hi = *reinterpret_cast<const v8bf*>(p + 16);
      afr[mf] = __builtin_shufflevector(lo, hi, 0, 1, 2, 3, 4, 5, 6, 7, 8, 9,
                                        10, 11, 12, 13, 14, 15);
    }
    v16bf bfr[2];
#pragma unroll
    for (int nf = 0; nf < 2; ++nf) {
      const unsigned short* p =
          &Bs[buf][(wn * 32 + nf * 16 + (lane & 15)) * LDP + kB];
      v8bf lo = *reinterpret_cast<const v8bf*>(p);
      v8bf hi = *reinterpret_cast<const v8bf*>(p + 16);
      bfr[nf] = __builtin_shufflevector(lo, hi, 0, 1, 2, 3, 4, 5, 6, 7, 8, 9,
                                        10, 11, 12, 13, 14, 15);
    }
#pragma unroll
    for (int mf = 0; mf < 4; ++mf)
#pragma unroll
      for (int nf = 0; nf < 2; ++nf)
        acc[mf][nf] = __builtin_amdgcn_wmma_f32_16x16x32_bf16(
            false, afr[mf], false, bfr[nf], (short)0, acc[mf][nf], false,
            false);
    asm volatile("s_wait_asynccnt 0" ::: "memory");
    __syncthreads();
  }
}

// GEMM1: h1 = relu(h @ fc1_w + fc1_b) * sigmoid(s*efc1[t])  -> bf16 [M][A]
__global__ void __launch_bounds__(256)
k_gemm_fc1(const unsigned short* __restrict__ A,
           const unsigned short* __restrict__ BT,
           const float* __restrict__ bias, const float* __restrict__ eg,
           const int* __restrict__ tptr, const int* __restrict__ sptr,
           unsigned short* __restrict__ out) {
  v8f acc[4][2] = {};
  gemm_staged<HH>(A, BT, acc);
  int lane = threadIdx.x & 31;
  int wave = threadIdx.x >> 5;
  int wm = wave >> 2, wn = wave & 3;
  int rowBase = blockIdx.y * 128 + wm * 64;
  int colBase = blockIdx.x * 128 + wn * 32;
  int   t  = *tptr;
  float sf = (float)(*sptr);
#pragma unroll
  for (int mf = 0; mf < 4; ++mf)
#pragma unroll
    for (int nf = 0; nf < 2; ++nf)
#pragma unroll
      for (int r = 0; r < 8; ++r) {
        int gr = rowBase + mf * 16 + r + ((lane >> 4) << 3);
        int gc = colBase + nf * 16 + (lane & 15);
        float v = acc[mf][nf][r] + bias[gc];
        v = fmaxf(v, 0.0f);
        v *= sigmoidf_(sf * eg[(long)t * AA + gc]);
        out[(long)gr * AA + gc] = f2bf(v);
      }
}

// GEMM2: out = x + relu(h1 @ fc2_w + fc2_b) * sigmoid(s*efc2[t])  -> f32
__global__ void __launch_bounds__(256)
k_gemm_fc2(const unsigned short* __restrict__ A,
           const unsigned short* __restrict__ BT,
           const float* __restrict__ bias, const float* __restrict__ eg,
           const int* __restrict__ tptr, const int* __restrict__ sptr,
           const float* __restrict__ x, float* __restrict__ out) {
  v8f acc[4][2] = {};
  gemm_staged<AA>(A, BT, acc);
  int lane = threadIdx.x & 31;
  int wave = threadIdx.x >> 5;
  int wm = wave >> 2, wn = wave & 3;
  int rowBase = blockIdx.y * 128 + wm * 64;
  int colBase = blockIdx.x * 128 + wn * 32;
  int   t  = *tptr;
  float sf = (float)(*sptr);
#pragma unroll
  for (int mf = 0; mf < 4; ++mf)
#pragma unroll
    for (int nf = 0; nf < 2; ++nf)
#pragma unroll
      for (int r = 0; r < 8; ++r) {
        int gr = rowBase + mf * 16 + r + ((lane >> 4) << 3);
        int gc = colBase + nf * 16 + (lane & 15);
        float v = acc[mf][nf][r] + bias[gc];
        v = fmaxf(v, 0.0f);
        v *= sigmoidf_(sf * eg[(long)t * HH + gc]);
        out[(long)gr * HH + gc] = x[(long)gr * HH + gc] + v;
      }
}

// ---------------------------------------------------------------------------
extern "C" void kernel_launch(void* const* d_in, const int* in_sizes, int n_in,
                              void* d_out, int out_size, void* d_ws,
                              size_t ws_size, hipStream_t stream) {
  const float* x        = (const float*)d_in[0];
  const float* fc1_w    = (const float*)d_in[1];
  const float* fc1_b    = (const float*)d_in[2];
  const float* fc2_w    = (const float*)d_in[3];
  const float* fc2_b    = (const float*)d_in[4];
  const float* efc1     = (const float*)d_in[5];
  const float* efc2     = (const float*)d_in[6];
  const float* sem_w    = (const float*)d_in[7];
  const float* sem_b    = (const float*)d_in[8];
  const float* route_w  = (const float*)d_in[9];
  const float* larger_w = (const float*)d_in[10];
  const float* larger_b = (const float*)d_in[11];
  const float* elarger  = (const float*)d_in[12];
  const int*   tptr     = (const int*)d_in[13];
  const int*   sptr     = (const int*)d_in[14];
  float* out = (float*)d_out;

  // workspace carving (256B aligned)
  char* ws = (char*)d_ws;
  size_t off = 0;
  auto take = [&](size_t bytes) {
    char* p = ws + off;
    off = (off + bytes + 255) & ~(size_t)255;
    return p;
  };
  unsigned short* semA  = (unsigned short*)take((size_t)NT * BB * SC * 2);
  unsigned short* rwT   = (unsigned short*)take((size_t)CC * NT * SS * SC * 2);
  unsigned short* fc1T  = (unsigned short*)take((size_t)AA * HH * 2);
  unsigned short* fc2T  = (unsigned short*)take((size_t)HH * AA * 2);
  float*          priors= (float*)take((size_t)CC * BB * NT * SS * 4);
  float*          vote  = (float*)take((size_t)CC * BB * SS * 4);
  unsigned short* hA    = (unsigned short*)take((size_t)MM * HH * 2);
  unsigned short* h1    = (unsigned short*)take((size_t)MM * AA * 2);
  (void)ws_size; (void)in_sizes; (void)n_in; (void)out_size;

  // 1) weight prep: transpose + bf16 convert
  k_transpose_bf16<<<4096, 256, 0, stream>>>(fc1_w, fc1T, 1, HH, AA);
  k_transpose_bf16<<<4096, 256, 0, stream>>>(fc2_w, fc2T, 1, AA, HH);
  k_transpose_bf16<<<4096, 256, 0, stream>>>(route_w, rwT, CC * NT, SC, SS);

  // 2) semantic capsules + squash
  k_sem_squash<<<MM / 256, 256, 0, stream>>>(x, sem_w, sem_b, semA);

  // 3) priors via WMMA: 1920 tiles, 4 waves/block
  k_priors_wmma<<<480, 128, 0, stream>>>(semA, rwT, priors);

  // 4) dynamic routing
  k_routing<<<CC * BB, SS, 0, stream>>>(priors, tptr, vote);

  // 5) h = x + gated capsule output  (bf16 A operand)
  k_build_h<<<(MM * HH) / 256, 256, 0, stream>>>(x, vote, larger_w, larger_b,
                                                 elarger, tptr, sptr, hA);

  // 6) adapter GEMM 1: [16384 x 768] x [768 x 2048]
  {
    dim3 grid(AA / 128, MM / 128);
    k_gemm_fc1<<<grid, 256, 0, stream>>>(hA, fc1T, fc1_b, efc1, tptr, sptr, h1);
  }
  // 7) adapter GEMM 2 + residual: [16384 x 2048] x [2048 x 768]
  {
    dim3 grid(HH / 128, MM / 128);
    k_gemm_fc2<<<grid, 256, 0, stream>>>(h1, fc2T, fc2_b, efc2, tptr, sptr, x,
                                         out);
  }
}
